// MultiheadAttention_90056874262679
// MI455X (gfx1250) — matrix operand order repaired and checked by reading, not compiled, not run on hardware
//
#include <hip/hip_runtime.h>
#include <cstdint>
#include <cstddef>

// ---------------------------------------------------------------------------
// MHA forward for MI455X (gfx1250, wave32, WMMA).
//   B=2, S=2048, D=1024, H=16, Hd=64.
// All matmuls: v_wmma_f32_16x16x32_bf16 (bf16 in, fp32 accumulate).
// Flash-attention schedule (no SxS materialization), double-buffered LDS,
// async global->LDS staging (ASYNCcnt) where the toolchain provides it.
// ---------------------------------------------------------------------------

static constexpr int BATCH = 2;
static constexpr int SEQ   = 2048;
static constexpr int DIM   = 1024;
static constexpr int NH    = 16;
static constexpr int HD    = 64;
static constexpr int MROWS = BATCH * SEQ;        // 4096
static constexpr float ATT_SCALE = 0.125f;       // 64^-0.5 (exact in bf16)

typedef __bf16 bf16;
typedef __attribute__((ext_vector_type(16))) __bf16 v16bf;
typedef __attribute__((ext_vector_type(8)))  __bf16 v8bf;
typedef __attribute__((ext_vector_type(8)))  float  v8f;

// ---- CDNA5 async global->LDS copy (ASYNCcnt path), feature-guarded --------
#if defined(__HIP_DEVICE_COMPILE__) && defined(__gfx1250__)
  #if __has_builtin(__builtin_amdgcn_global_load_async_to_lds_b128) && \
      __has_builtin(__builtin_amdgcn_s_wait_asynccnt)
    #define USE_ASYNC_LDS 1
  #endif
#endif
#ifndef USE_ASYNC_LDS
#define USE_ASYNC_LDS 0
#endif

#if USE_ASYNC_LDS
// Builtin signature (from hipcc diagnostic): pointers to a 16-byte int vector,
// in address spaces 1 (global) and 3 (LDS), plus two constant ints.
typedef int v4i_vs __attribute__((vector_size(16)));
typedef __attribute__((address_space(1))) v4i_vs* as1_v4i_p;
typedef __attribute__((address_space(3))) v4i_vs* as3_v4i_p;

__device__ __forceinline__ void async_cp_b128(const void* g, void* l) {
  __builtin_amdgcn_global_load_async_to_lds_b128(
      (as1_v4i_p)(g), (as3_v4i_p)(l), 0, 0);
}
__device__ __forceinline__ void async_wait0() {
  __builtin_amdgcn_s_wait_asynccnt(0);
}
#endif

// ---- helpers --------------------------------------------------------------

__device__ __forceinline__ uint16_t f2bfbits(float f) {
  union { float f; uint32_t u; } a; a.f = f;
  uint32_t u = a.u + 0x7FFFu + ((a.u >> 16) & 1u);   // round-to-nearest-even
  return (uint16_t)(u >> 16);
}
__device__ __forceinline__ bf16 f2bf(float f) {
  union { uint16_t u; bf16 h; } r; r.u = f2bfbits(f);
  return r.h;
}

// Build one v16bf fragment from two 16-byte-aligned 8-element chunks.
__device__ __forceinline__ v16bf ld_frag_pair(const bf16* p0, const bf16* p1) {
  v8bf a = *reinterpret_cast<const v8bf*>(p0);
  v8bf b = *reinterpret_cast<const v8bf*>(p1);
  return __builtin_shufflevector(a, b, 0,1,2,3,4,5,6,7,8,9,10,11,12,13,14,15);
}

__device__ __forceinline__ v8f wmma_bf16(v16bf a, v16bf b, v8f c) {
#if defined(__gfx1250__)
  // D = A(16x32) x B(32x16) + C, fp32 accumulate
  return __builtin_amdgcn_wmma_f32_16x16x32_bf16(false, a, false, b,
                                                 (short)0, c, false, false);
#else
  return c;
#endif
}

// ---- fp32 -> bf16 cast (vectorized x4) ------------------------------------

__global__ __launch_bounds__(256) void cast_f32_bf16_x4(
    const float* __restrict__ s, bf16* __restrict__ d, int n4) {
  int i = blockIdx.x * blockDim.x + threadIdx.x;
  if (i >= n4) return;
  float4 f = reinterpret_cast<const float4*>(s)[i];
  uint32_t lo = (uint32_t)f2bfbits(f.x) | ((uint32_t)f2bfbits(f.y) << 16);
  uint32_t hi = (uint32_t)f2bfbits(f.z) | ((uint32_t)f2bfbits(f.w) << 16);
  reinterpret_cast<uint2*>(d)[i] = make_uint2(lo, hi);
}

// ---- WMMA GEMM:  Y = oscale * X (M x 1024) * W^T --------------------------
// 128x128 block tile, BK=32, double-buffered LDS, 8 waves ->
// each wave owns 2x4 16x16 WMMA accumulators (8 v_wmma / k-step, 32 k-steps).

enum { MODE_HEADS = 0,    // store bf16 head-split [b,h,s,hd]   (Q, K)
       MODE_HEADS_T = 1,  // store bf16 transposed [b,h,hd,s]   (V^T)
       MODE_F32 = 2 };    // store fp32 [m,n]                   (final proj)

template<int MODE>
__global__ __launch_bounds__(256) void gemm_xwT(
    const bf16* __restrict__ Ag, const bf16* __restrict__ Wg,
    void* __restrict__ Yv, float oscale) {
  __shared__ __align__(16) bf16 As[2][128][40];   // +8 pad: bank relief
  __shared__ __align__(16) bf16 Bs[2][128][40];   // 40 KB total

  const int tid  = threadIdx.x;
  const int lane = tid & 31;
  const int w    = tid >> 5;       // wave 0..7 (wave32)
  const int wm   = w & 3;          // m-subtiles 2*wm .. 2*wm+1
  const int wn   = w >> 2;         // n-subtiles 4*wn .. 4*wn+3
  const int hs   = lane >> 4;      // lane half select (ISA A/B layouts)
  const int ln   = lane & 15;

  const int m0 = blockIdx.y * 128;
  const int n0 = blockIdx.x * 128;

  const int lrow = tid >> 1;          // 0..127 (tile staging row)
  const int lcol = (tid & 1) * 16;    // 0 or 16

  v8f acc[2][4];
  #pragma unroll
  for (int mi = 0; mi < 2; mi++)
    #pragma unroll
    for (int ni = 0; ni < 4; ni++)
      #pragma unroll
      for (int j = 0; j < 8; j++) acc[mi][ni][j] = 0.f;

  const bf16* aRow = Ag + (size_t)(m0 + lrow) * DIM;
  const bf16* bRow = Wg + (size_t)(n0 + lrow) * DIM;
  constexpr int NIT = DIM / 32;

#if USE_ASYNC_LDS
  // Prologue: DMA tile 0 straight into LDS (no VGPR staging).
  async_cp_b128(aRow + lcol,     &As[0][lrow][lcol]);
  async_cp_b128(aRow + lcol + 8, &As[0][lrow][lcol + 8]);
  async_cp_b128(bRow + lcol,     &Bs[0][lrow][lcol]);
  async_cp_b128(bRow + lcol + 8, &Bs[0][lrow][lcol + 8]);
  async_wait0();
  __syncthreads();
#else
  uint4 a0, a1, b0, b1;
  {
    const uint4* ap = reinterpret_cast<const uint4*>(aRow + lcol);
    const uint4* bp = reinterpret_cast<const uint4*>(bRow + lcol);
    a0 = ap[0]; a1 = ap[1]; b0 = bp[0]; b1 = bp[1];
    *reinterpret_cast<uint4*>(&As[0][lrow][lcol])     = a0;
    *reinterpret_cast<uint4*>(&As[0][lrow][lcol + 8]) = a1;
    *reinterpret_cast<uint4*>(&Bs[0][lrow][lcol])     = b0;
    *reinterpret_cast<uint4*>(&Bs[0][lrow][lcol + 8]) = b1;
  }
  __syncthreads();
#endif

  for (int it = 0; it < NIT; ++it) {
    const int cur  = it & 1, nxt = cur ^ 1;
    const int k0   = it * 32;
    const bool more = (it + 1) < NIT;

#if USE_ASYNC_LDS
    // Kick next tile's DMA before doing any math on the current one.
    if (more) {
      async_cp_b128(aRow + k0 + 32 + lcol,     &As[nxt][lrow][lcol]);
      async_cp_b128(aRow + k0 + 32 + lcol + 8, &As[nxt][lrow][lcol + 8]);
      async_cp_b128(bRow + k0 + 32 + lcol,     &Bs[nxt][lrow][lcol]);
      async_cp_b128(bRow + k0 + 32 + lcol + 8, &Bs[nxt][lrow][lcol + 8]);
    }
#else
    if (more) {
      const uint4* ap = reinterpret_cast<const uint4*>(aRow + k0 + 32 + lcol);
      const uint4* bp = reinterpret_cast<const uint4*>(bRow + k0 + 32 + lcol);
      a0 = ap[0]; a1 = ap[1]; b0 = bp[0]; b1 = bp[1];
    }
#endif
    if ((it + 2) < NIT) {   // -> global_prefetch_b8 for tile after next
      __builtin_prefetch(aRow + k0 + 64 + lcol, 0, 0);
      __builtin_prefetch(bRow + k0 + 64 + lcol, 0, 0);
    }

    // ---- 8 WMMAs on buffer [cur] ----
    v16bf afrag[2], bfrag[4];
    #pragma unroll
    for (int mi = 0; mi < 2; mi++) {
      // A layout: lane m holds row m; K = hs*16 .. hs*16+15 contiguous
      const bf16* r = &As[cur][(wm * 2 + mi) * 16 + ln][0];
      afrag[mi] = ld_frag_pair(r + hs * 16, r + hs * 16 + 8);
    }
    #pragma unroll
    for (int ni = 0; ni < 4; ni++) {
      // B layout: lane n holds column n; K = hs*16 .. hs*16+15 contiguous
      const bf16* r = &Bs[cur][(wn * 4 + ni) * 16 + ln][0];
      bfrag[ni] = ld_frag_pair(r + hs * 16, r + hs * 16 + 8);
    }
    #pragma unroll
    for (int mi = 0; mi < 2; mi++)
      #pragma unroll
      for (int ni = 0; ni < 4; ni++)
        acc[mi][ni] = wmma_bf16(afrag[mi], bfrag[ni], acc[mi][ni]);

#if USE_ASYNC_LDS
    if (more) async_wait0();   // ASYNCcnt not covered by __syncthreads
#else
    if (more) {
      *reinterpret_cast<uint4*>(&As[nxt][lrow][lcol])     = a0;
      *reinterpret_cast<uint4*>(&As[nxt][lrow][lcol + 8]) = a1;
      *reinterpret_cast<uint4*>(&Bs[nxt][lrow][lcol])     = b0;
      *reinterpret_cast<uint4*>(&Bs[nxt][lrow][lcol + 8]) = b1;
    }
#endif
    __syncthreads();
  }

  // C/D layout: component j, lane l -> row = j + 8*(l>>4), col = l&15
  #pragma unroll
  for (int mi = 0; mi < 2; mi++) {
    #pragma unroll
    for (int ni = 0; ni < 4; ni++) {
      #pragma unroll
      for (int j = 0; j < 8; j++) {
        const int row = m0 + (wm * 2 + mi) * 16 + j + 8 * hs;
        const int col = n0 + (wn * 4 + ni) * 16 + ln;
        const float v = acc[mi][ni][j] * oscale;
        if (MODE == MODE_F32) {
          reinterpret_cast<float*>(Yv)[(size_t)row * DIM + col] = v;
        } else {
          const int b = row / SEQ, s = row % SEQ;
          const int h = col / HD,  d = col % HD;
          if (MODE == MODE_HEADS) {
            reinterpret_cast<bf16*>(Yv)[(((size_t)(b * NH + h) * SEQ) + s) * HD + d] = f2bf(v);
          } else { // MODE_HEADS_T : V^T [b,h,d,s]
            reinterpret_cast<bf16*>(Yv)[(((size_t)(b * NH + h) * HD) + d) * SEQ + s] = f2bf(v);
          }
        }
      }
    }
  }
}

// ---- Flash attention (causal) ---------------------------------------------
// Grid: (SEQ/128, BATCH*NH). Block: 256 threads = 8 waves.
// Each wave owns 16 query rows (Q frags pinned in registers).
// KV tiles of 64 keys double-buffered in LDS.
// Per tile / wave: 8 WMMA (QK^T) + 8 WMMA (PV). Q pre-scaled by 1/sqrt(d).

__global__ __launch_bounds__(256) void flash_attn_wmma(
    const bf16* __restrict__ Qg, const bf16* __restrict__ Kg,
    const bf16* __restrict__ Vtg, bf16* __restrict__ Cg) {
  __shared__ __align__(16) bf16 Ks[2][64][72];    // K tile   [key][d]
  __shared__ __align__(16) bf16 Vs[2][64][72];    // V^T tile [d][key]
  __shared__ __align__(16) bf16 Ps[8][16][72];    // per-wave P (C->A relayout)

  const int tid  = threadIdx.x;
  const int lane = tid & 31;
  const int w    = tid >> 5;
  const int hs   = lane >> 4;
  const int ln   = lane & 15;

  const int bh = blockIdx.y;          // 0..31
  const int b  = bh / NH;
  const int h  = bh % NH;
  const int r0 = blockIdx.x * 128;
  const int qrow0 = r0 + w * 16;

  const bf16* Qb = Qg  + (size_t)bh * SEQ * HD;
  const bf16* Kb = Kg  + (size_t)bh * SEQ * HD;
  const bf16* Vb = Vtg + (size_t)bh * HD * SEQ;

  // Q fragments (16 rows x 64) live in registers for the whole kernel.
  v16bf qf[2];
  {
    const bf16* qr = Qb + (size_t)(qrow0 + ln) * HD;
    #pragma unroll
    for (int ks = 0; ks < 2; ks++)
      qf[ks] = ld_frag_pair(qr + ks * 32 + hs * 16, qr + ks * 32 + hs * 16 + 8);
  }

  v8f oacc[4];
  #pragma unroll
  for (int ni = 0; ni < 4; ni++)
    #pragma unroll
    for (int j = 0; j < 8; j++) oacc[ni][j] = 0.f;

  float mrow[8], lsum[8];
  #pragma unroll
  for (int j = 0; j < 8; j++) { mrow[j] = -1e30f; lsum[j] = 0.f; }

  // KV staging (register double-buffer of the LDS tiles)
  const int rr = tid >> 2;               // 0..63
  const int cc = (tid & 3) * 16;         // 0,16,32,48
  uint4 kv0r, kv1r, vv0r, vv1r;
  auto loadKV = [&](int kv0) {
    const uint4* kp = reinterpret_cast<const uint4*>(Kb + (size_t)(kv0 + rr) * HD + cc);
    const uint4* vp = reinterpret_cast<const uint4*>(Vb + (size_t)rr * SEQ + kv0 + cc);
    kv0r = kp[0]; kv1r = kp[1]; vv0r = vp[0]; vv1r = vp[1];
  };
  auto storeKV = [&](int buf) {
    *reinterpret_cast<uint4*>(&Ks[buf][rr][cc])     = kv0r;
    *reinterpret_cast<uint4*>(&Ks[buf][rr][cc + 8]) = kv1r;
    *reinterpret_cast<uint4*>(&Vs[buf][rr][cc])     = vv0r;
    *reinterpret_cast<uint4*>(&Vs[buf][rr][cc + 8]) = vv1r;
  };

  const int kv_end = min(SEQ, r0 + 128);   // causal upper bound for block
  const int nT = kv_end >> 6;              // 64-key tiles
  loadKV(0);
  storeKV(0);

  for (int it = 0; it < nT; ++it) {
    const int cur = it & 1, nxt = cur ^ 1;
    const int kv0 = it << 6;
    __syncthreads();                       // buffer [cur] ready for everyone
    if (it + 1 < nT) loadKV(kv0 + 64);     // overlap next tile's global traffic

    // scores S = Q K^T : B-fragment column n = key row (contiguous along d)
    v8f sc[4];
    #pragma unroll
    for (int ni = 0; ni < 4; ni++)
      #pragma unroll
      for (int j = 0; j < 8; j++) sc[ni][j] = 0.f;
    #pragma unroll
    for (int ni = 0; ni < 4; ni++) {
      #pragma unroll
      for (int ks = 0; ks < 2; ks++) {
        const bf16* kr = &Ks[cur][ni * 16 + ln][ks * 32 + hs * 16];
        v16bf bfr = ld_frag_pair(kr, kr + 8);
        sc[ni] = wmma_bf16(qf[ks], bfr, sc[ni]);
      }
    }

    // causal mask (C-layout: row = qrow0 + j + 8*hs, col = kv0 + ni*16 + ln)
    const int rbase = qrow0 + 8 * hs;
    #pragma unroll
    for (int ni = 0; ni < 4; ni++) {
      const int col = kv0 + ni * 16 + ln;
      #pragma unroll
      for (int j = 0; j < 8; j++)
        if (col > rbase + j) sc[ni][j] = -1e30f;
    }

    // online softmax: 16-lane xor reductions handle both row-halves at once
    #pragma unroll
    for (int j = 0; j < 8; j++) {
      float rmax = fmaxf(fmaxf(sc[0][j], sc[1][j]), fmaxf(sc[2][j], sc[3][j]));
      #pragma unroll
      for (int off = 1; off < 16; off <<= 1)
        rmax = fmaxf(rmax, __shfl_xor(rmax, off, 32));
      const float nm    = fmaxf(mrow[j], rmax);
      const float alpha = __expf(mrow[j] - nm);
      mrow[j] = nm;
      float rs = 0.f;
      #pragma unroll
      for (int ni = 0; ni < 4; ni++) {
        float p = __expf(sc[ni][j] - nm);
        sc[ni][j] = p;
        rs += p;
      }
      #pragma unroll
      for (int off = 1; off < 16; off <<= 1)
        rs += __shfl_xor(rs, off, 32);
      lsum[j] = lsum[j] * alpha + rs;
      #pragma unroll
      for (int ni = 0; ni < 4; ni++) oacc[ni][j] *= alpha;
    }

    // relayout P (C-layout regs) -> A-layout via per-wave LDS region
    #pragma unroll
    for (int ni = 0; ni < 4; ni++)
      #pragma unroll
      for (int j = 0; j < 8; j++)
        Ps[w][j + 8 * hs][ni * 16 + ln] = f2bf(sc[ni][j]);
    __syncthreads();

    // O += P V : B-fragment column n = output dim d (contiguous keys in V^T)
    #pragma unroll
    for (int ks = 0; ks < 2; ks++) {
      const bf16* pr = &Ps[w][ln][0];
      v16bf pa = ld_frag_pair(pr + ks * 32 + hs * 16, pr + ks * 32 + hs * 16 + 8);
      #pragma unroll
      for (int ni = 0; ni < 4; ni++) {
        const bf16* vr = &Vs[cur][ni * 16 + ln][ks * 32 + hs * 16];
        v16bf vb = ld_frag_pair(vr, vr + 8);
        oacc[ni] = wmma_bf16(pa, vb, oacc[ni]);
      }
    }

    if (it + 1 < nT) storeKV(nxt);   // fill ping-pong buffer for next round
  }

  // finalize: divide by row sum, merge heads -> ctx [b,s, h*64+d] bf16
  #pragma unroll
  for (int ni = 0; ni < 4; ni++) {
    #pragma unroll
    for (int j = 0; j < 8; j++) {
      const int row = qrow0 + j + 8 * hs;
      const int col = ni * 16 + ln;
      const float o = oacc[ni][j] / lsum[j];
      Cg[((size_t)(b * SEQ + row)) * DIM + h * HD + col] = f2bf(o);
    }
  }
}

// ---- launcher -------------------------------------------------------------

extern "C" void kernel_launch(void* const* d_in, const int* in_sizes, int n_in,
                              void* d_out, int out_size, void* d_ws, size_t ws_size,
                              hipStream_t stream) {
  const float* x  = (const float*)d_in[0];
  const float* Wq = (const float*)d_in[1];
  const float* Wk = (const float*)d_in[2];
  const float* Wv = (const float*)d_in[3];
  const float* Wo = (const float*)d_in[4];

  char* ws = (char*)d_ws;
  size_t off = 0;
  auto carve = [&](size_t bytes) -> char* {
    char* p = ws + off;
    off += (bytes + 255) & ~(size_t)255;
    return p;
  };
  bf16* xb  = (bf16*)carve((size_t)MROWS * DIM * sizeof(bf16));
  bf16* wqb = (bf16*)carve((size_t)DIM * DIM * sizeof(bf16));
  bf16* wkb = (bf16*)carve((size_t)DIM * DIM * sizeof(bf16));
  bf16* wvb = (bf16*)carve((size_t)DIM * DIM * sizeof(bf16));
  bf16* wob = (bf16*)carve((size_t)DIM * DIM * sizeof(bf16));
  bf16* qb  = (bf16*)carve((size_t)MROWS * DIM * sizeof(bf16));
  bf16* kb  = (bf16*)carve((size_t)MROWS * DIM * sizeof(bf16));
  bf16* vtb = (bf16*)carve((size_t)MROWS * DIM * sizeof(bf16));
  bf16* cb  = (bf16*)carve((size_t)MROWS * DIM * sizeof(bf16));
  (void)ws_size; (void)in_sizes; (void)n_in; (void)out_size;   // ~48 MB total

  // fp32 -> bf16 casts
  const int n4x = MROWS * DIM / 4;
  const int n4w = DIM * DIM / 4;
  cast_f32_bf16_x4<<<(n4x + 255) / 256, 256, 0, stream>>>(x,  xb,  n4x);
  cast_f32_bf16_x4<<<(n4w + 255) / 256, 256, 0, stream>>>(Wq, wqb, n4w);
  cast_f32_bf16_x4<<<(n4w + 255) / 256, 256, 0, stream>>>(Wk, wkb, n4w);
  cast_f32_bf16_x4<<<(n4w + 255) / 256, 256, 0, stream>>>(Wv, wvb, n4w);
  cast_f32_bf16_x4<<<(n4w + 255) / 256, 256, 0, stream>>>(Wo, wob, n4w);

  // Q/K/V projections. Q pre-scaled by 1/sqrt(Hd) (exact power of two).
  // V stored transposed per-head for the PV matmul.
  const dim3 gg(DIM / 128, MROWS / 128);   // (8, 32)
  gemm_xwT<MODE_HEADS>  <<<gg, 256, 0, stream>>>(xb, wqb, qb,  ATT_SCALE);
  gemm_xwT<MODE_HEADS>  <<<gg, 256, 0, stream>>>(xb, wkb, kb,  1.0f);
  gemm_xwT<MODE_HEADS_T><<<gg, 256, 0, stream>>>(xb, wvb, vtb, 1.0f);

  // causal flash attention
  flash_attn_wmma<<<dim3(SEQ / 128, BATCH * NH), 256, 0, stream>>>(qb, kb, vtb, cb);

  // output projection -> fp32 d_out
  gemm_xwT<MODE_F32><<<gg, 256, 0, stream>>>(cb, wob, (void*)d_out, 1.0f);
}
